// Shift_tcn_76373108457575
// MI455X (gfx1250) — matrix operand order repaired and checked by reading, not compiled, not run on hardware
//
#include <hip/hip_runtime.h>
#include <hip/hip_bf16.h>
#include <stdint.h>

// ---------------- problem dimensions (fixed by setup_inputs) ----------------
#define NN   64
#define CC   256          // Cin == Cout
#define TT   64
#define VV   25
#define TVSZ (TT * VV)    // 1600
#define NTVS (NN * TVSZ)  // 102400
#define EPSF 1e-5f

// LDS tile: 64 j-rows x 256 K bf16, row pitch padded 512 -> 528 bytes
#define BROWS   64
#define BPITCH  528

// ---------------- CDNA5 WMMA / TDM vector types ----------------
typedef __attribute__((ext_vector_type(16))) __bf16          v16bf;
typedef __attribute__((ext_vector_type(8)))  float           v8f;
typedef __attribute__((ext_vector_type(8)))  unsigned short  v8us;
typedef __attribute__((ext_vector_type(16))) unsigned short  v16us;
typedef __attribute__((ext_vector_type(4)))  unsigned int    u32x4;
typedef __attribute__((ext_vector_type(4)))  int             i32x4;
typedef __attribute__((ext_vector_type(8)))  int             i32x8;

#if __has_include(<hip/amd_detail/amd_gfx1250_TDM.h>)
#define TDM_SIX_ARGS 1
#endif

__device__ __forceinline__ unsigned short f2bf(float f) {
    unsigned u = __builtin_bit_cast(unsigned, f);
    u += 0x7FFFu + ((u >> 16) & 1u);        // round-to-nearest-even
    return (unsigned short)(u >> 16);
}

__device__ __forceinline__ v16bf combine_bf(v8us lo, v8us hi) {
    v16us w = __builtin_shufflevector(lo, hi,
        0,1,2,3,4,5,6,7,8,9,10,11,12,13,14,15);
    return __builtin_bit_cast(v16bf, w);
}

// ---------------- kernel 1: BN1 per-channel stats ----------------
__global__ void bn1_stats_kernel(const float* __restrict__ x,
                                 float* __restrict__ mean1,
                                 float* __restrict__ rstd1) {
    const int c   = blockIdx.x;
    const int tid = threadIdx.x;
    const float* xc = x + (size_t)c * TVSZ;
    float s = 0.f, q = 0.f;
    for (int n = 0; n < NN; ++n) {
        const float* xr = xc + (size_t)n * (CC * TVSZ);
        for (int i = tid; i < TVSZ; i += 256) {
            float v = xr[i];
            s += v; q += v * v;
        }
    }
    __shared__ float sh_s[256], sh_q[256];
    sh_s[tid] = s; sh_q[tid] = q;
    __syncthreads();
    for (int off = 128; off > 0; off >>= 1) {
        if (tid < off) { sh_s[tid] += sh_s[tid + off]; sh_q[tid] += sh_q[tid + off]; }
        __syncthreads();
    }
    if (tid == 0) {
        float m   = sh_s[0] * (1.0f / (float)NTVS);
        float var = sh_q[0] * (1.0f / (float)NTVS) - m * m;
        mean1[c] = m;
        rstd1[c] = rsqrtf(var + EPSF);
    }
}

// ---------------- kernel 2: pack conv weight to bf16 (row-major o x c) ------
__global__ void wpack_kernel(const float* __restrict__ w,
                             unsigned short* __restrict__ Wbf) {
    int i = blockIdx.x * 256 + threadIdx.x;
    if (i < CC * CC) Wbf[i] = f2bf(w[i]);
}

// ------- kernel 3: BN1 apply + temporal shift(theta_in) + pack H[j][c] bf16 -
__global__ void bn1_shift_pack_kernel(const float* __restrict__ x,
                                      const float* __restrict__ mean1,
                                      const float* __restrict__ rstd1,
                                      const float* __restrict__ g1,
                                      const float* __restrict__ b1,
                                      const float* __restrict__ theta,
                                      unsigned short* __restrict__ Hbf) {
    const int t   = blockIdx.x;
    const int n   = blockIdx.y;
    const int tid = threadIdx.x;
    __shared__ float hs[CC * VV];   // 25600 B

    for (int base = 0; base < CC * 32; base += 256) {
        int e  = base + tid;
        int c  = e >> 5;
        int vi = e & 31;
        if (vi < VV) {
            float th = theta[c];
            float fl = floorf(th);
            int   d0 = (int)fl;
            float fr = th - fl;
            float m  = mean1[c], r = rstd1[c], gg = g1[c], bb = b1[c];
            int t0 = t + d0, t1 = t0 + 1;
            const float* xc = x + (size_t)n * (CC * TVSZ) + (size_t)c * TVSZ;
            float s0 = 0.f, s1 = 0.f;
            if (t0 >= 0 && t0 < TT) s0 = (xc[t0 * VV + vi] - m) * r * gg + bb;
            if (t1 >= 0 && t1 < TT) s1 = (xc[t1 * VV + vi] - m) * r * gg + bb;
            hs[c * VV + vi] = (1.f - fr) * s0 + fr * s1;
        }
    }
    __syncthreads();
    unsigned short* hrow = Hbf + ((size_t)(n * TT + t) * VV) * CC;
    const int c = tid;
    #pragma unroll 5
    for (int v = 0; v < VV; ++v)
        hrow[(size_t)v * CC + c] = f2bf(hs[c * VV + v]);   // coalesced in c
}

// ---------------- kernel 4: WMMA GEMM  P[o][j] = relu(W x H + b) ------------
// 512 threads = 16 waves; wave w owns o-tile w. Block-shared B tile (64 j
// columns x K=256 bf16 = 32KB) is DMA'd into LDS by the Tensor Data Mover
// with a padded row pitch (528B) so per-lane ds_load_b128 fetches are
// bank-conflict free. Each wave: 4 j-tiles x 8 K-steps = 32 v_wmma.
__global__ void __launch_bounds__(512)
gemm_wmma_kernel(const unsigned short* __restrict__ Wbf,
                 const unsigned short* __restrict__ Hbf,
                 const float* __restrict__ bias,
                 float* __restrict__ P) {
    const int tid  = threadIdx.x;
    const int wave = tid >> 5;
    const int lane = tid & 31;
    const bool lohalf = lane < 16;
    const int mrow = lane & 15;
    const int ot   = wave;               // o-tile index 0..15
    const int j0   = blockIdx.x * 64;    // first of 64 j columns

    __shared__ __attribute__((aligned(16))) unsigned char smemB[BROWS * BPITCH];

    // ---- stage B tile into LDS ----
#if __has_builtin(__builtin_amdgcn_tensor_load_to_lds)
    if (tid < 32) {
        unsigned lds_off = (unsigned)(uintptr_t)(&smemB[0]);
        unsigned long long gaddr =
            (unsigned long long)(uintptr_t)(Hbf + (size_t)j0 * CC);
        // ---- D# group 0: count=1, lds_addr, global_addr, type=2 ----
        u32x4 g0;
        g0[0] = 1u;                                  // count=1, user mode
        g0[1] = lds_off;                             // lds_addr [63:32]
        g0[2] = (unsigned)(gaddr & 0xFFFFFFFFull);   // global_addr lo
        g0[3] = (unsigned)((gaddr >> 32) & 0x01FFFFFFull) | 0x80000000u; // type=2
        // ---- D# group 1: data_size=4B(code2), pad 4 DW per 128 DW row ----
        unsigned td0 = 128;                 // dwords per j-row (256 bf16)
        unsigned td1 = (unsigned)(NTVS - j0);
        unsigned tile0 = 128, tile1 = BROWS, tile2 = 0;
        unsigned long long st0 = 128;       // row stride in dwords
        i32x8 g1v;
        g1v[0] = (int)((2u << 16) | (1u << 20) | (6u << 22) | (3u << 25));
        g1v[1] = (int)((td0 & 0xFFFFu) << 16);
        g1v[2] = (int)(((td0 >> 16) & 0xFFFFu) | ((td1 & 0xFFFFu) << 16));
        g1v[3] = (int)(((td1 >> 16) & 0xFFFFu) | (tile0 << 16));
        g1v[4] = (int)(tile1 | (tile2 << 16));
        g1v[5] = (int)(st0 & 0xFFFFFFFFull);
        g1v[6] = (int)((st0 >> 32) & 0xFFFFu);   // + tensor_dim1_stride lo16 = 0
        g1v[7] = 0;
        i32x4 gz = {};                      // groups 2/3 unused (2D tensor)
#if defined(TDM_SIX_ARGS)
        i32x8 gz8 = {};
        __builtin_amdgcn_tensor_load_to_lds(g0, g1v, gz, gz, gz8, 0);
#else
        __builtin_amdgcn_tensor_load_to_lds(g0, g1v, gz, gz, 0);
#endif
        __builtin_amdgcn_s_wait_tensorcnt(0);
    }
#else
    // fallback: cooperative global->LDS copy with the same 528B pitch
    for (int idx = tid; idx < BROWS * 32; idx += 512) {
        int row = idx >> 5, seg = idx & 31;
        *(v8us*)(&smemB[row * BPITCH + seg * 16]) =
            *(const v8us*)(Hbf + (size_t)(j0 + row) * CC + seg * 8);
    }
#endif
    __syncthreads();

    v8f acc0 = {}, acc1 = {}, acc2 = {}, acc3 = {};

    // A operand: lane holds K (lane<16 ? 0..15 : 16..31), contiguous
    const unsigned short* arow =
        Wbf + (size_t)(ot * 16 + mrow) * CC + (lohalf ? 0 : 16);
    // B operand base in LDS: lane holds column (lane&15) of its j-tile,
    // K = (lane<16 ? 0..15 : 16..31) + kt*32  -> contiguous 32B per lane
    const unsigned char* brow0 = &smemB[mrow * BPITCH + (lohalf ? 0 : 32)];

    #pragma unroll
    for (int kt = 0; kt < 8; ++kt) {
        const unsigned short* ap = arow + kt * 32;
        v16bf a = combine_bf(*(const v8us*)ap, *(const v8us*)(ap + 8));

        const unsigned char* bp = brow0 + kt * 64;
        v16bf b0 = combine_bf(*(const v8us*)(bp),
                              *(const v8us*)(bp + 16));
        v16bf b1 = combine_bf(*(const v8us*)(bp + 16 * BPITCH),
                              *(const v8us*)(bp + 16 * BPITCH + 16));
        v16bf b2 = combine_bf(*(const v8us*)(bp + 32 * BPITCH),
                              *(const v8us*)(bp + 32 * BPITCH + 16));
        v16bf b3 = combine_bf(*(const v8us*)(bp + 48 * BPITCH),
                              *(const v8us*)(bp + 48 * BPITCH + 16));

        acc0 = __builtin_amdgcn_wmma_f32_16x16x32_bf16(false, a, false, b0, (short)0, acc0, false, false);
        acc1 = __builtin_amdgcn_wmma_f32_16x16x32_bf16(false, a, false, b1, (short)0, acc1, false, false);
        acc2 = __builtin_amdgcn_wmma_f32_16x16x32_bf16(false, a, false, b2, (short)0, acc2, false, false);
        acc3 = __builtin_amdgcn_wmma_f32_16x16x32_bf16(false, a, false, b3, (short)0, acc3, false, false);
    }

    // D layout: VGPR r is row (r + (lane<16?0:8)), column (lane&15).
    const int obase = ot * 16 + (lohalf ? 0 : 8);
    const int j_    = j0 + mrow;
    #pragma unroll
    for (int r = 0; r < 8; ++r) {
        int o_ = obase + r;
        float bb = bias[o_];
        size_t row = (size_t)o_ * NTVS;
        P[row + j_]      = fmaxf(acc0[r] + bb, 0.f);
        P[row + j_ + 16] = fmaxf(acc1[r] + bb, 0.f);
        P[row + j_ + 32] = fmaxf(acc2[r] + bb, 0.f);
        P[row + j_ + 48] = fmaxf(acc3[r] + bb, 0.f);
    }
}

// ---------------- kernel 5: BN2 stats over shifted(relu(conv)) --------------
__global__ void bn2_stats_kernel(const float* __restrict__ P,
                                 const float* __restrict__ theta,
                                 float* __restrict__ mean2,
                                 float* __restrict__ rstd2) {
    const int o   = blockIdx.x;
    const int tid = threadIdx.x;
    float th = theta[o];
    float fl = floorf(th);
    int   d0 = (int)fl;
    float fr = th - fl;
    const float* Po = P + (size_t)o * NTVS;
    float s = 0.f, q = 0.f;
    for (int n = 0; n < NN; ++n) {
        const float* Pn = Po + (size_t)n * TVSZ;
        for (int i = tid; i < TVSZ; i += 256) {
            int t = i / VV, v = i - t * VV;
            int t0 = t + d0, t1 = t0 + 1;
            float a = (t0 >= 0 && t0 < TT) ? Pn[t0 * VV + v] : 0.f;
            float b = (t1 >= 0 && t1 < TT) ? Pn[t1 * VV + v] : 0.f;
            float val = (1.f - fr) * a + fr * b;
            s += val; q += val * val;
        }
    }
    __shared__ float sh_s[256], sh_q[256];
    sh_s[tid] = s; sh_q[tid] = q;
    __syncthreads();
    for (int off = 128; off > 0; off >>= 1) {
        if (tid < off) { sh_s[tid] += sh_s[tid + off]; sh_q[tid] += sh_q[tid + off]; }
        __syncthreads();
    }
    if (tid == 0) {
        float m   = sh_s[0] * (1.0f / (float)NTVS);
        float var = sh_q[0] * (1.0f / (float)NTVS) - m * m;
        mean2[o] = m;
        rstd2[o] = rsqrtf(var + EPSF);
    }
}

// ---------------- kernel 6: final shift + BN2 apply -> out[n][o][t][v] ------
__global__ void bn2_apply_kernel(const float* __restrict__ P,
                                 const float* __restrict__ theta,
                                 const float* __restrict__ mean2,
                                 const float* __restrict__ rstd2,
                                 const float* __restrict__ g2,
                                 const float* __restrict__ b2,
                                 float* __restrict__ out) {
    const int o   = blockIdx.x;
    const int n   = blockIdx.y;
    const int tid = threadIdx.x;
    float th = theta[o];
    float fl = floorf(th);
    int   d0 = (int)fl;
    float fr = th - fl;
    float m  = mean2[o], r = rstd2[o], gg = g2[o], bb = b2[o];
    const float* Pn = P + (size_t)o * NTVS + (size_t)n * TVSZ;
    float* on = out + ((size_t)n * CC + o) * TVSZ;
    for (int i = tid; i < TVSZ; i += 256) {
        int t = i / VV, v = i - t * VV;
        int t0 = t + d0, t1 = t0 + 1;
        float a = (t0 >= 0 && t0 < TT) ? Pn[t0 * VV + v] : 0.f;
        float b = (t1 >= 0 && t1 < TT) ? Pn[t1 * VV + v] : 0.f;
        float val = (1.f - fr) * a + fr * b;
        on[i] = (val - m) * r * gg + bb;
    }
}

// ---------------- launcher --------------------------------------------------
extern "C" void kernel_launch(void* const* d_in, const int* in_sizes, int n_in,
                              void* d_out, int out_size, void* d_ws, size_t ws_size,
                              hipStream_t stream) {
    const float* x         = (const float*)d_in[0];
    const float* conv_w    = (const float*)d_in[1];
    const float* conv_b    = (const float*)d_in[2];
    const float* bn1_g     = (const float*)d_in[3];
    const float* bn1_b     = (const float*)d_in[4];
    const float* bn2_g     = (const float*)d_in[5];
    const float* bn2_b     = (const float*)d_in[6];
    const float* theta_in  = (const float*)d_in[7];
    const float* theta_out = (const float*)d_in[8];
    float* out = (float*)d_out;

    // workspace carve-up (all offsets 256B-aligned)
    char* ws = (char*)d_ws;
    size_t offH = 0;                                   // H bf16: [NTV][C]
    size_t offW = offH + (size_t)NTVS * CC * 2;        // 52,428,800
    size_t offP = offW + (size_t)CC * CC * 2;          // +131,072
    size_t offS = offP + (size_t)CC * NTVS * 4;        // +104,857,600
    unsigned short* Hbf = (unsigned short*)(ws + offH);
    unsigned short* Wbf = (unsigned short*)(ws + offW);
    float* P     = (float*)(ws + offP);                // [Cout][NTV]
    float* mean1 = (float*)(ws + offS);
    float* rstd1 = mean1 + CC;
    float* mean2 = rstd1 + CC;
    float* rstd2 = mean2 + CC;
    (void)ws_size; (void)in_sizes; (void)n_in; (void)out_size;

    bn1_stats_kernel<<<CC, 256, 0, stream>>>(x, mean1, rstd1);
    wpack_kernel<<<(CC * CC + 255) / 256, 256, 0, stream>>>(conv_w, Wbf);
    bn1_shift_pack_kernel<<<dim3(TT, NN), 256, 0, stream>>>(
        x, mean1, rstd1, bn1_g, bn1_b, theta_in, Hbf);
    gemm_wmma_kernel<<<NTVS / 64, 512, 0, stream>>>(Wbf, Hbf, conv_b, P);
    bn2_stats_kernel<<<CC, 256, 0, stream>>>(P, theta_out, mean2, rstd2);
    bn2_apply_kernel<<<dim3(CC, NN), 256, 0, stream>>>(
        P, theta_out, mean2, rstd2, bn2_g, bn2_b, out);
}
